// TimeMix_48026324304454
// MI455X (gfx1250) — matrix operand order repaired and checked by reading, not compiled, not run on hardware
//
#include <hip/hip_runtime.h>
#include <math.h>

// ---------------- problem constants (from reference) ----------------
constexpr int B_ = 4;
constexpr int T_ = 2048;
constexpr int C_ = 768;
constexpr int H_ = 12;
constexpr int N_ = 64;
constexpr int M_ = B_ * T_;          // 8192 rows for all GEMMs
constexpr float EPS_GN = 0.00064f;

typedef __attribute__((ext_vector_type(16))) __bf16 v16bf;
typedef __attribute__((ext_vector_type(8)))  __bf16 v8bf;
typedef __attribute__((ext_vector_type(8)))  float  v8f;
typedef unsigned int v4u __attribute__((ext_vector_type(4)));
typedef int          v4i __attribute__((ext_vector_type(4)));
typedef int          v8i __attribute__((ext_vector_type(8)));

__device__ __forceinline__ float sigf(float x) { return 1.0f / (1.0f + expf(-x)); }

// ---------------- token-shift mix prep: 6 bf16 outputs ----------------
__global__ void prep_kernel(const float* __restrict__ x,
                            const float* __restrict__ lr, const float* __restrict__ lw,
                            const float* __restrict__ lk, const float* __restrict__ lv,
                            const float* __restrict__ la, const float* __restrict__ lg,
                            __bf16* __restrict__ xr, __bf16* __restrict__ xw,
                            __bf16* __restrict__ xk, __bf16* __restrict__ xv,
                            __bf16* __restrict__ xa, __bf16* __restrict__ xg) {
  int idx = blockIdx.x * blockDim.x + threadIdx.x;
  if (idx >= M_ * C_) return;
  int c = idx % C_;
  int t = (idx / C_) % T_;
  float xc = x[idx];
  float xs = (t == 0) ? 0.0f : x[idx - C_];   // ZeroPad2d((0,0,1,-1))
  float xd = xs - xc;
  xr[idx] = (__bf16)(xc + xd * lr[c]);
  xw[idx] = (__bf16)(xc + xd * lw[c]);
  xk[idx] = (__bf16)(xc + xd * lk[c]);
  xv[idx] = (__bf16)(xc + xd * lv[c]);
  xa[idx] = (__bf16)(xc + xd * la[c]);
  xg[idx] = (__bf16)(xc + xd * lg[c]);
}

// ---------------- f32 -> bf16 helpers ----------------
__global__ void cvt_bf16_kernel(const float* __restrict__ s, __bf16* __restrict__ d, int n) {
  int i = blockIdx.x * blockDim.x + threadIdx.x;
  if (i < n) d[i] = (__bf16)s[i];
}
__global__ void tanh_bf16_kernel(const float* __restrict__ s, __bf16* __restrict__ d, int n) {
  int i = blockIdx.x * blockDim.x + threadIdx.x;
  if (i < n) d[i] = (__bf16)tanhf(s[i]);
}
__global__ void sig_bf16_kernel(const float* __restrict__ s, __bf16* __restrict__ d, int n) {
  int i = blockIdx.x * blockDim.x + threadIdx.x;
  if (i < n) d[i] = (__bf16)sigf(s[i]);
}
// transpose K x N (f32) -> N x K (bf16) so GEMM operand is K-contiguous
__global__ void tr_cvt_kernel(const float* __restrict__ s, __bf16* __restrict__ d, int K, int N) {
  int k = blockIdx.x * 16 + threadIdx.x;
  int n = blockIdx.y * 16 + threadIdx.y;
  if (k < K && n < N) d[(size_t)n * K + k] = (__bf16)s[(size_t)k * N + n];
}

// ---------------- common WMMA fragment helpers ----------------
// ISA 16-bit A layout: lane half 0 -> K {0..7, 16..23}; half 1 -> K {8..15, 24..31}
__device__ __forceinline__ v16bf load_a_frag(const __bf16* p) {
  v8bf lo = *reinterpret_cast<const v8bf*>(p);
  v8bf hi = *reinterpret_cast<const v8bf*>(p + 16);
  return __builtin_shufflevector(lo, hi, 0, 1, 2, 3, 4, 5, 6, 7,
                                 8, 9, 10, 11, 12, 13, 14, 15);
}

// ---------------- direct-load bf16 WMMA GEMM (small GEMMs) ----------------
// C[m,n] = sum_k A[m,k]*W[n,k]. A: MxK bf16, W: NxK bf16, C: MxN f32.
// 128 threads = 4 waves; block tile 64x64; wave tile 32x32 (4 accums).
__global__ __launch_bounds__(128) void gemm_bf16_wmma(const __bf16* __restrict__ A,
                                                      const __bf16* __restrict__ W,
                                                      float* __restrict__ Cmat,
                                                      int M, int N, int K) {
  const int lane = threadIdx.x & 31;
  const int wave = threadIdx.x >> 5;
  const int half = lane >> 4;
  const int l15  = lane & 15;
  const int m0 = blockIdx.x * 64 + (wave >> 1) * 32;
  const int n0 = blockIdx.y * 64 + (wave & 1) * 32;

  v8f acc00 = {}, acc01 = {}, acc10 = {}, acc11 = {};
  const __bf16* a0p = A + (size_t)(m0 + l15) * K + half * 8;
  const __bf16* a1p = A + (size_t)(m0 + 16 + l15) * K + half * 8;
  // B uses same K order as A: lane half 0 -> K {0..7, 16..23}; half 1 -> K {8..15, 24..31}
  const __bf16* b0p = W + (size_t)(n0 + l15) * K + half * 8;
  const __bf16* b1p = W + (size_t)(n0 + 16 + l15) * K + half * 8;

  for (int k0 = 0; k0 < K; k0 += 32) {
    v16bf a0 = load_a_frag(a0p + k0);
    v16bf a1 = load_a_frag(a1p + k0);
    v16bf b0 = load_a_frag(b0p + k0);
    v16bf b1 = load_a_frag(b1p + k0);
    acc00 = __builtin_amdgcn_wmma_f32_16x16x32_bf16(false, a0, false, b0, (short)0, acc00, false, false);
    acc01 = __builtin_amdgcn_wmma_f32_16x16x32_bf16(false, a0, false, b1, (short)0, acc01, false, false);
    acc10 = __builtin_amdgcn_wmma_f32_16x16x32_bf16(false, a1, false, b0, (short)0, acc10, false, false);
    acc11 = __builtin_amdgcn_wmma_f32_16x16x32_bf16(false, a1, false, b1, (short)0, acc11, false, false);
  }
  const int nn = n0 + l15;
  #pragma unroll
  for (int i = 0; i < 8; ++i) {
    int mr = m0 + i + half * 8;
    Cmat[(size_t)mr * N + nn]             = acc00[i];
    Cmat[(size_t)mr * N + nn + 16]        = acc01[i];
    Cmat[(size_t)(mr + 16) * N + nn]      = acc10[i];
    Cmat[(size_t)(mr + 16) * N + nn + 16] = acc11[i];
  }
}

// ---------------- TDM-staged bf16 WMMA GEMM (big C x C GEMMs) ----------------
// Issues TENSOR_LOAD_TO_LDS (D# per ISA ch.8) to DMA 64x32 bf16 tiles of A and W
// into double-buffered LDS; waves consume via ds loads + WMMA; sync = TENSORcnt + barrier.
__device__ __forceinline__ void tdm_load_2d(const void* gaddr, unsigned lds_off_bytes,
                                            unsigned tensor_d0, unsigned tensor_d1,
                                            unsigned tile_d0, unsigned tile_d1,
                                            unsigned stride0) {
  unsigned long long ga = (unsigned long long)gaddr;
  v4u g0;
  g0[0] = 1u;                                                 // count=1, user descriptor
  g0[1] = lds_off_bytes;                                      // lds_addr (bytes)
  g0[2] = (unsigned)(ga & 0xffffffffu);                       // global_addr[31:0]
  g0[3] = (unsigned)((ga >> 32) & 0x01ffffffu) | (2u << 30);  // global_addr[56:32], type=2
  v8i g1;
  g1[0] = (int)(1u << 16);                                    // workgroup_mask=0, data_size=1 (2B)
  g1[1] = (int)((tensor_d0 & 0xffffu) << 16);                 // tensor_dim0[15:0]
  g1[2] = (int)(((tensor_d0 >> 16) & 0xffffu) | ((tensor_d1 & 0xffffu) << 16));
  g1[3] = (int)(((tensor_d1 >> 16) & 0xffffu) | ((tile_d0 & 0xffffu) << 16));
  g1[4] = (int)(tile_d1 & 0xffffu);                           // tile_dim1; tile_dim2=0 (2-D)
  g1[5] = (int)stride0;                                       // tensor_dim0_stride[31:0]
  g1[6] = 0;                                                  // stride0[47:32], stride1[15:0]
  g1[7] = 0;
  v4i z4 = {0, 0, 0, 0};
#if defined(__clang_major__) && (__clang_major__ >= 23)
  v8i z8 = {0, 0, 0, 0, 0, 0, 0, 0};
  __builtin_amdgcn_tensor_load_to_lds(g0, g1, z4, z4, z8, 0);
#else
  __builtin_amdgcn_tensor_load_to_lds(g0, g1, z4, z4, 0);
#endif
}

__global__ __launch_bounds__(128) void gemm_bf16_wmma_tdm(const __bf16* __restrict__ A,
                                                          const __bf16* __restrict__ W,
                                                          float* __restrict__ Cmat,
                                                          int M, int N, int K) {
  extern __shared__ __bf16 lds[];                 // 2 buffers x (A tile 64x32 + B tile 64x32)
  constexpr int TILE_E = 64 * 32;                 // elements per tile
  const int lane = threadIdx.x & 31;
  const int wave = threadIdx.x >> 5;
  const int half = lane >> 4;
  const int l15  = lane & 15;
  const int m0 = blockIdx.x * 64;
  const int n0 = blockIdx.y * 64;
  const int mloc = (wave >> 1) * 32;
  const int nloc = (wave & 1) * 32;

  v8f acc00 = {}, acc01 = {}, acc10 = {}, acc11 = {};

  auto issue = [&](int buf, int k0) {
    unsigned base = (unsigned)buf * (2u * TILE_E * 2u);       // bytes
    tdm_load_2d(A + (size_t)m0 * K + k0, base,                K, (unsigned)M, 32, 64, K);
    tdm_load_2d(W + (size_t)n0 * K + k0, base + TILE_E * 2u,  K, (unsigned)N, 32, 64, K);
  };

  if (wave == 0) issue(0, 0);
  const int nsteps = K / 32;
  for (int s = 0; s < nsteps; ++s) {
    const int buf = s & 1;
    if (wave == 0) {
      if (s + 1 < nsteps) {
        issue(buf ^ 1, (s + 1) * 32);
        __builtin_amdgcn_s_wait_tensorcnt(2);   // current buffer's pair retired (in-order)
      } else {
        __builtin_amdgcn_s_wait_tensorcnt(0);
      }
    }
    __syncthreads();
    const __bf16* aT = lds + (size_t)buf * 2 * TILE_E;
    const __bf16* bT = aT + TILE_E;
    v16bf a0 = load_a_frag(aT + (mloc + l15) * 32 + half * 8);
    v16bf a1 = load_a_frag(aT + (mloc + 16 + l15) * 32 + half * 8);
    v16bf b0 = load_a_frag(bT + (nloc + l15) * 32 + half * 8);
    v16bf b1 = load_a_frag(bT + (nloc + 16 + l15) * 32 + half * 8);
    acc00 = __builtin_amdgcn_wmma_f32_16x16x32_bf16(false, a0, false, b0, (short)0, acc00, false, false);
    acc01 = __builtin_amdgcn_wmma_f32_16x16x32_bf16(false, a0, false, b1, (short)0, acc01, false, false);
    acc10 = __builtin_amdgcn_wmma_f32_16x16x32_bf16(false, a1, false, b0, (short)0, acc10, false, false);
    acc11 = __builtin_amdgcn_wmma_f32_16x16x32_bf16(false, a1, false, b1, (short)0, acc11, false, false);
    __syncthreads();                           // everyone done reading buf before it is re-filled
  }
  const int nn = n0 + nloc + l15;
  #pragma unroll
  for (int i = 0; i < 8; ++i) {
    int mr = m0 + mloc + i + half * 8;
    Cmat[(size_t)mr * N + nn]             = acc00[i];
    Cmat[(size_t)mr * N + nn + 16]        = acc01[i];
    Cmat[(size_t)(mr + 16) * N + nn]      = acc10[i];
    Cmat[(size_t)(mr + 16) * N + nn + 16] = acc11[i];
  }
}

// ---------------- LoRA-path epilogues ----------------
__global__ void w_final_kernel(const float* __restrict__ p, const float* __restrict__ w_miu,
                               float* __restrict__ dec, int n) {
  int i = blockIdx.x * blockDim.x + threadIdx.x;
  if (i >= n) return;
  float u = w_miu[i % C_] + p[i];
  float w = -log1pf(expf(-u)) - 0.5f;     // -softplus(-u) - 0.5
  dec[i] = expf(-expf(w));                // decay used by the recurrence
}
__global__ void a_final_kernel(const float* __restrict__ p, const float* __restrict__ a_miu,
                               float* __restrict__ a, int n) {
  int i = blockIdx.x * blockDim.x + threadIdx.x;
  if (i < n) a[i] = sigf(a_miu[i % C_] + p[i]);
}
__global__ void v_final_kernel(const float* __restrict__ p, const float* __restrict__ v_miu,
                               const float* __restrict__ v_first, float* __restrict__ v, int n) {
  int i = blockIdx.x * blockDim.x + threadIdx.x;
  if (i >= n) return;
  float v0 = v[i];
  v[i] = v0 + (v_first[i] - v0) * sigf(v_miu[i % C_] + p[i]);
}

// ---------------- per-head k/kk post-processing ----------------
__global__ __launch_bounds__(32) void k_post_kernel(float* __restrict__ k, float* __restrict__ a_bb,
                                                    const float* __restrict__ k_k,
                                                    const float* __restrict__ k_a,
                                                    float* __restrict__ kk) {
  int h = blockIdx.x % H_;
  size_t base = (size_t)(blockIdx.x / H_) * C_ + h * N_;
  int lane = threadIdx.x;
  int c0 = h * N_ + lane, c1 = c0 + 32;
  size_t i0 = base + lane, i1 = i0 + 32;
  float kx0 = k[i0], kx1 = k[i1];
  float av0 = a_bb[i0], av1 = a_bb[i1];
  float kr0 = kx0 * k_k[c0], kr1 = kx1 * k_k[c1];
  float ss = kr0 * kr0 + kr1 * kr1;
  for (int off = 16; off; off >>= 1) ss += __shfl_xor(ss, off, 32);
  float inv = 1.0f / fmaxf(sqrtf(ss), 1e-12f);
  float kk0 = kr0 * inv, kk1 = kr1 * inv;
  kk[i0] = kk0;  kk[i1] = kk1;
  k[i0] = kx0 * (1.0f + (av0 - 1.0f) * k_a[c0]);
  k[i1] = kx1 * (1.0f + (av1 - 1.0f) * k_a[c1]);
  a_bb[i0] = kk0 * av0;  a_bb[i1] = kk1 * av1;
}

// ---------------- WKV-7 recurrence: one block per (b,h), thread i owns row i ----------------
// Per-step vectors staged in LDS; {decay,b,k,r} packed as float4 -> single ds_load_b128 per j.
__global__ __launch_bounds__(64) void wkv7_kernel(const float* __restrict__ r, const float* __restrict__ d,
                                                  const float* __restrict__ k, const float* __restrict__ kk,
                                                  const float* __restrict__ bb, const float* __restrict__ v,
                                                  float* __restrict__ y) {
  int b = blockIdx.x / H_;
  int h = blockIdx.x % H_;
  int i = threadIdx.x;
  __shared__ float4 q[N_];     // {d, b, k, r}
  __shared__ float  sz[N_];
  __shared__ float  sv[N_];
  float S[N_];
  #pragma unroll
  for (int j = 0; j < N_; ++j) S[j] = 0.0f;
  size_t base = (size_t)b * T_ * C_ + h * N_;
  for (int t = 0; t < T_; ++t, base += C_) {
    size_t idx = base + i;
    q[i] = make_float4(d[idx], bb[idx], k[idx], r[idx]);
    sz[i] = -kk[idx];
    sv[i] = v[idx];
    __syncthreads();
    float sa = 0.0f;
    #pragma unroll
    for (int j = 0; j < N_; ++j) sa += S[j] * sz[j];
    float vi = sv[i];
    float yv = 0.0f;
    #pragma unroll
    for (int j = 0; j < N_; ++j) {
      float4 qq = q[j];
      float s = S[j] * qq.x + sa * qq.y + vi * qq.z;
      S[j] = s;
      yv += s * qq.w;
    }
    y[idx] = yv;
    __syncthreads();
  }
}

// ---------------- GroupNorm + bonus + gate -> bf16 for W_o GEMM ----------------
__global__ __launch_bounds__(32) void out_prep_kernel(const float* __restrict__ y, const float* __restrict__ r,
                                                      const float* __restrict__ k, const float* __restrict__ v,
                                                      const float* __restrict__ g, const float* __restrict__ r_k,
                                                      const float* __restrict__ ln_w, const float* __restrict__ ln_b,
                                                      __bf16* __restrict__ ym) {
  int h = blockIdx.x % H_;
  size_t base = (size_t)(blockIdx.x / H_) * C_ + h * N_;
  int lane = threadIdx.x;
  int c0 = h * N_ + lane, c1 = c0 + 32;
  size_t i0 = base + lane, i1 = i0 + 32;
  float y0 = y[i0], y1 = y[i1];
  float sum = y0 + y1;
  float sq  = y0 * y0 + y1 * y1;
  float dp  = r[i0] * k[i0] * r_k[c0] + r[i1] * k[i1] * r_k[c1];
  for (int off = 16; off; off >>= 1) {
    sum += __shfl_xor(sum, off, 32);
    sq  += __shfl_xor(sq,  off, 32);
    dp  += __shfl_xor(dp,  off, 32);
  }
  float mu   = sum * (1.0f / 64.0f);
  float var  = sq * (1.0f / 64.0f) - mu * mu;   // biased variance (torch)
  float rstd = rsqrtf(var + EPS_GN);
  float o0 = (y0 - mu) * rstd * ln_w[c0] + ln_b[c0] + dp * v[i0];
  float o1 = (y1 - mu) * rstd * ln_w[c1] + ln_b[c1] + dp * v[i1];
  ym[i0] = (__bf16)(o0 * g[i0]);
  ym[i1] = (__bf16)(o1 * g[i1]);
}

// ---------------- host-side orchestration ----------------
extern "C" void kernel_launch(void* const* d_in, const int* in_sizes, int n_in,
                              void* d_out, int out_size, void* d_ws, size_t ws_size,
                              hipStream_t stream) {
  (void)in_sizes; (void)n_in; (void)out_size; (void)ws_size;
  const float* x       = (const float*)d_in[0];
  const float* v_first = (const float*)d_in[1];
  const float* lr   = (const float*)d_in[2];
  const float* lw   = (const float*)d_in[3];
  const float* lk   = (const float*)d_in[4];
  const float* lv   = (const float*)d_in[5];
  const float* la   = (const float*)d_in[6];
  const float* lg   = (const float*)d_in[7];
  const float* w_miu = (const float*)d_in[8];
  const float* w_A  = (const float*)d_in[9];
  const float* w_Bm = (const float*)d_in[10];
  const float* a_miu = (const float*)d_in[11];
  const float* a_A  = (const float*)d_in[12];
  const float* a_Bm = (const float*)d_in[13];
  const float* v_miu = (const float*)d_in[14];
  const float* v_A  = (const float*)d_in[15];
  const float* v_Bm = (const float*)d_in[16];
  const float* g_A  = (const float*)d_in[17];
  const float* g_Bm = (const float*)d_in[18];
  const float* k_k  = (const float*)d_in[19];
  const float* k_a  = (const float*)d_in[20];
  const float* r_k  = (const float*)d_in[21];
  const float* W_r  = (const float*)d_in[22];
  const float* W_k  = (const float*)d_in[23];
  const float* W_v  = (const float*)d_in[24];
  const float* W_o  = (const float*)d_in[25];
  const float* ln_w = (const float*)d_in[26];
  const float* ln_b = (const float*)d_in[27];
  float* out = (float*)d_out;

  const size_t MC = (size_t)M_ * C_;
  char* ws = (char*)d_ws;
  size_t off = 0;
  auto alloc = [&](size_t bytes) -> void* {
    void* p = ws + off;
    off = (off + bytes + 255) & ~(size_t)255;
    return p;
  };
  // bf16 activations
  __bf16* xr_b = (__bf16*)alloc(MC * 2);
  __bf16* xw_b = (__bf16*)alloc(MC * 2);
  __bf16* xk_b = (__bf16*)alloc(MC * 2);
  __bf16* xv_b = (__bf16*)alloc(MC * 2);
  __bf16* xa_b = (__bf16*)alloc(MC * 2);
  __bf16* xg_b = (__bf16*)alloc(MC * 2);
  __bf16* sb   = (__bf16*)alloc(MC * 2);      // shared bf16 scratch (lora temps, v0b, ym)
  // f32 activations
  float* r_f  = (float*)alloc(MC * 4);
  float* k_f  = (float*)alloc(MC * 4);
  float* v_f  = (float*)alloc(MC * 4);
  float* d_f  = (float*)alloc(MC * 4);        // decay = exp(-exp(w))
  float* a_f  = (float*)alloc(MC * 4);        // a, later b = kk*a
  float* kk_f = (float*)alloc(MC * 4);
  float* g_f  = (float*)alloc(MC * 4);
  float* p_f  = (float*)alloc(MC * 4);        // lora pre-act scratch, later y
  float* t64  = (float*)alloc((size_t)M_ * 64 * 4);
  float* t128 = (float*)alloc((size_t)M_ * 128 * 4);
  // bf16 weights
  __bf16* Wr_b = (__bf16*)alloc((size_t)C_ * C_ * 2);
  __bf16* Wk_b = (__bf16*)alloc((size_t)C_ * C_ * 2);
  __bf16* Wv_b = (__bf16*)alloc((size_t)C_ * C_ * 2);
  __bf16* Wo_b = (__bf16*)alloc((size_t)C_ * C_ * 2);
  __bf16* wA_t = (__bf16*)alloc((size_t)C_ * 64 * 2);
  __bf16* wB_t = (__bf16*)alloc((size_t)C_ * 64 * 2);
  __bf16* aA_t = (__bf16*)alloc((size_t)C_ * 64 * 2);
  __bf16* aB_t = (__bf16*)alloc((size_t)C_ * 64 * 2);
  __bf16* vA_t = (__bf16*)alloc((size_t)C_ * 64 * 2);
  __bf16* vB_t = (__bf16*)alloc((size_t)C_ * 64 * 2);
  __bf16* gA_t = (__bf16*)alloc((size_t)C_ * 128 * 2);
  __bf16* gB_t = (__bf16*)alloc((size_t)C_ * 128 * 2);

  const int EW = 256;
  auto blocks = [&](size_t n) { return (unsigned)((n + EW - 1) / EW); };
  auto gemm = [&](const __bf16* A, const __bf16* W, float* Cm, int M, int N, int K) {
    gemm_bf16_wmma<<<dim3(M / 64, N / 64), 128, 0, stream>>>(A, W, Cm, M, N, K);
  };
  auto gemm_tdm = [&](const __bf16* A, const __bf16* W, float* Cm, int M, int N, int K) {
    gemm_bf16_wmma_tdm<<<dim3(M / 64, N / 64), 128, 2 * 2 * 64 * 32 * 2, stream>>>(A, W, Cm, M, N, K);
  };
  auto trc = [&](const float* s, __bf16* d, int K, int N) {
    tr_cvt_kernel<<<dim3((K + 15) / 16, (N + 15) / 16), dim3(16, 16), 0, stream>>>(s, d, K, N);
  };

  // weight conversion (W_* used as x @ W.T -> already N-major/K-contiguous)
  cvt_bf16_kernel<<<blocks((size_t)C_ * C_), EW, 0, stream>>>(W_r, Wr_b, C_ * C_);
  cvt_bf16_kernel<<<blocks((size_t)C_ * C_), EW, 0, stream>>>(W_k, Wk_b, C_ * C_);
  cvt_bf16_kernel<<<blocks((size_t)C_ * C_), EW, 0, stream>>>(W_v, Wv_b, C_ * C_);
  cvt_bf16_kernel<<<blocks((size_t)C_ * C_), EW, 0, stream>>>(W_o, Wo_b, C_ * C_);
  trc(w_A, wA_t, C_, 64);   trc(w_Bm, wB_t, 64, C_);
  trc(a_A, aA_t, C_, 64);   trc(a_Bm, aB_t, 64, C_);
  trc(v_A, vA_t, C_, 64);   trc(v_Bm, vB_t, 64, C_);
  trc(g_A, gA_t, C_, 128);  trc(g_Bm, gB_t, 128, C_);

  // token-shift mixes
  prep_kernel<<<blocks(MC), EW, 0, stream>>>(x, lr, lw, lk, lv, la, lg,
                                             xr_b, xw_b, xk_b, xv_b, xa_b, xg_b);

  // big projections (TDM-staged bf16 WMMA, f32 accumulate)
  gemm_tdm(xr_b, Wr_b, r_f, M_, C_, C_);
  gemm_tdm(xk_b, Wk_b, k_f, M_, C_, C_);
  gemm_tdm(xv_b, Wv_b, v_f, M_, C_, C_);

  // w path: w = -softplus(-(w_miu + tanh(xw@w_A)@w_B)) - 0.5 ; decay = exp(-exp(w))
  gemm(xw_b, wA_t, t64, M_, 64, C_);
  tanh_bf16_kernel<<<blocks((size_t)M_ * 64), EW, 0, stream>>>(t64, sb, M_ * 64);
  gemm(sb, wB_t, p_f, M_, C_, 64);
  w_final_kernel<<<blocks(MC), EW, 0, stream>>>(p_f, w_miu, d_f, (int)MC);

  // a path: a = sigmoid(a_miu + (xa@a_A)@a_B)
  gemm(xa_b, aA_t, t64, M_, 64, C_);
  cvt_bf16_kernel<<<blocks((size_t)M_ * 64), EW, 0, stream>>>(t64, sb, M_ * 64);
  gemm(sb, aB_t, p_f, M_, C_, 64);
  a_final_kernel<<<blocks(MC), EW, 0, stream>>>(p_f, a_miu, a_f, (int)MC);

  // v path: v = v0 + (v_first - v0)*sigmoid(v_miu + (v0@v_A)@v_B)
  cvt_bf16_kernel<<<blocks(MC), EW, 0, stream>>>(v_f, sb, (int)MC);
  gemm(sb, vA_t, t64, M_, 64, C_);
  cvt_bf16_kernel<<<blocks((size_t)M_ * 64), EW, 0, stream>>>(t64, sb, M_ * 64);
  gemm(sb, vB_t, p_f, M_, C_, 64);
  v_final_kernel<<<blocks(MC), EW, 0, stream>>>(p_f, v_miu, v_first, v_f, (int)MC);

  // g path: g = sigmoid(xg@g_A)@g_B
  gemm(xg_b, gA_t, t128, M_, 128, C_);
  sig_bf16_kernel<<<blocks((size_t)M_ * 128), EW, 0, stream>>>(t128, sb, M_ * 128);
  gemm(sb, gB_t, g_f, M_, C_, 128);

  // per-head normalize kk, adjust k, fold b = kk*a
  k_post_kernel<<<M_ * H_, 32, 0, stream>>>(k_f, a_f, k_k, k_a, kk_f);

  // sequential recurrence -> y (reuses p_f)
  wkv7_kernel<<<B_ * H_, 64, 0, stream>>>(r_f, d_f, k_f, kk_f, a_f, v_f, p_f);

  // groupnorm + bonus + gate -> bf16
  out_prep_kernel<<<M_ * H_, 32, 0, stream>>>(p_f, r_f, k_f, v_f, g_f, r_k, ln_w, ln_b, sb);

  // out = ym @ W_o.T  (written straight to d_out)
  gemm_tdm(sb, Wo_b, out, M_, C_, C_);

  // second tuple element: v_first passthrough (LAYER_ID != 0)
  hipMemcpyAsync(out + MC, v_first, MC * sizeof(float), hipMemcpyDeviceToDevice, stream);
}
